// RoPEAttention_36318243455183
// MI455X (gfx1250) — compile-verified, matrix operand order checked by reading, not run on hardware
//
#include <hip/hip_runtime.h>

// ---------------------------------------------------------------------------
// RoPE multi-head attention for MI455X (gfx1250, wave32, WMMA).
// All matmuls via V_WMMA_F32_16X16X32_F16 (f16 in, f32 accumulate).
// GEMMs: 64x64 per-wave tiles, shared A slab staged to LDS with
// global_load_async_to_lds_b128 (double-buffered, s_wait_asynccnt).
// Attention: causal flash loop, no S x S materialization.
// ---------------------------------------------------------------------------

#define EMB       2048
#define N_HEADS   16
#define HEAD_DIM  128
#define N_KV      4
#define BB        2
#define SS        2048
#define TOK       (BB * SS)              // 4096 tokens
#define QDIM      (N_HEADS * HEAD_DIM)   // 2048
#define KVDIM     (N_KV * HEAD_DIM)      // 512

typedef __attribute__((ext_vector_type(16))) _Float16 v16h;
typedef __attribute__((ext_vector_type(8)))  _Float16 v8h;
typedef __attribute__((ext_vector_type(8)))  float    v8f;

// ---------------------------------------------------------------------------
// Fragment helpers (ISA 7.12.2).
// A-fragment 16x32 f16: lane L holds row M = L&15;
//   lanes 0-15 : K = {0..7, 16..23},  lanes 16-31 : K = {8..15, 24..31}
// B-fragment uses the same per-lane K striping with column N = L&15, so a
// B operand stored as B^T row-major ([N,K]) loads identically to A.
// Both halves are contiguous 16B runs -> two b128 loads per fragment.
// ---------------------------------------------------------------------------
__device__ __forceinline__ v16h frag_rowmajor(const _Float16* __restrict__ p,
                                              long ld, long r0, long k0) {
  const int  lane = threadIdx.x & 31;
  const long r = r0 + (lane & 15);
  const long k = k0 + ((long)(lane >> 4) << 3);
  const _Float16* q = p + r * ld + k;
  union { v16h v; v8h h[2]; } u;
  u.h[0] = *(const v8h*)(q);        // K run  k .. k+7
  u.h[1] = *(const v8h*)(q + 16);   // K run  k+16 .. k+23
  return u.v;
}

__device__ __forceinline__ v8f wmma16(v16h a, v16h b, v8f c) {
  return __builtin_amdgcn_wmma_f32_16x16x32_f16(
      /*neg_a=*/false, a, /*neg_b=*/false, b,
      /*c_mod=*/(short)0, c, /*reuse_a=*/false, /*reuse_b=*/false);
}

// Async global -> LDS copy (CDNA5, ASYNCcnt-tracked, no VGPR round-trip).
// VDST VGPR carries the LDS byte offset (generic LDS addr truncates to [31:0]).
__device__ __forceinline__ void async_copy_b128(unsigned lds_off,
                                                const void* gaddr) {
  asm volatile("global_load_async_to_lds_b128 %0, %1, off"
               :: "v"(lds_off), "v"((unsigned long long)gaddr) : "memory");
}
__device__ __forceinline__ void wait_async0() {
  asm volatile("s_wait_asynccnt 0x0" ::: "memory");
}

// ---------------------------------------------------------------------------
// Elementwise conversions
// ---------------------------------------------------------------------------
__global__ void cvt_f32_to_f16(const float* __restrict__ in,
                               _Float16* __restrict__ out, long n) {
  long i = (long)blockIdx.x * blockDim.x + threadIdx.x;
  if (i < n) out[i] = (_Float16)in[i];
}

// in: [K, N] f32 row-major  ->  out: [N, K] f16 row-major (B^T for WMMA)
__global__ void transpose_cvt(const float* __restrict__ in,
                              _Float16* __restrict__ out, long K, long N) {
  long i = (long)blockIdx.x * blockDim.x + threadIdx.x;
  if (i < N * K) {
    long n = i / K, k = i % K;
    out[i] = (_Float16)in[k * N + n];
  }
}

// ---------------------------------------------------------------------------
// WMMA GEMM: C[M,N] = A[M,K] (f16 row-major) x B (given as B^T [N,K]).
// Block = 8 waves, each owning a 64x64 tile; the 8 waves share one 64-row A
// slab per 32-wide k-step, async-staged into double-buffered LDS.
// Requires (N/64) % 8 == 0 and grid exactly (M/64)*(N/64)/8 blocks.
// MODE 0: f16 row-major out, MODE 1: f16 transposed out ([N,M]), MODE 2: f32.
// ---------------------------------------------------------------------------
template <int MODE>
__global__ __launch_bounds__(256) void gemm_wmma(
    const _Float16* __restrict__ A, const _Float16* __restrict__ Bt,
    void* __restrict__ Cout, int M, int N, int K) {
  const int wave_in_blk = threadIdx.x >> 5;
  const int lane = threadIdx.x & 31;
  const int blk_per_row = (N >> 6) >> 3;          // blocks per 64-row stripe
  const long wm = (long)(blockIdx.x / blk_per_row) * 64;   // block-uniform
  const long wn = ((long)(blockIdx.x % blk_per_row) * 8 + wave_in_blk) * 64;

  // 64 rows x 32 halves, padded to 40 halves (80B rows -> 20r mod 64 banks
  // are all distinct for the 16-lane fragment reads), double-buffered.
  __shared__ _Float16 ldsA[2][64][40];

  // Each of the 256 threads async-copies one 16B chunk per k-step:
  // thread t -> row t/4, 8-half chunk t%4.
  const int srow = threadIdx.x >> 2;
  const int schunk = threadIdx.x & 3;
  const unsigned lds_off[2] = {
      (unsigned)(size_t)&ldsA[0][srow][schunk * 8],
      (unsigned)(size_t)&ldsA[1][srow][schunk * 8] };
  const _Float16* gA = A + (wm + srow) * (long)K + schunk * 8;

  v8f acc[4][4] = {};

  async_copy_b128(lds_off[0], gA);                // prologue: stage k=0 slab
  for (int k = 0; k < K; k += 32) {
    const int p = (k >> 5) & 1;
    wait_async0();                 // this wave's slab contribution landed
    __syncthreads();               // whole slab ready; prior reads of buf p^1
                                   // were consumed before each wave's barrier
    if (k + 32 < K)                // overlap next slab with current compute
      async_copy_b128(lds_off[p ^ 1], gA + (k + 32));

    const _Float16* As = &ldsA[p][0][0];
    v16h af[4], bf[4];
#pragma unroll
    for (int i = 0; i < 4; ++i) af[i] = frag_rowmajor(As, 40, i * 16, 0);
#pragma unroll
    for (int j = 0; j < 4; ++j) {
      bf[j] = frag_rowmajor(Bt, K, wn + j * 16, k);
      if (k + 32 < K)              // warm L0/L2 for next k-step's B fragment
        __builtin_prefetch(Bt + (wn + j * 16 + (lane & 15)) * (long)K +
                           (k + 32) + ((lane >> 4) << 3), 0, 3);
    }
#pragma unroll
    for (int i = 0; i < 4; ++i)
#pragma unroll
      for (int j = 0; j < 4; ++j)
        acc[i][j] = wmma16(af[i], bf[j], acc[i][j]);
  }

  // C layout (ISA 7.12.2): elem e -> row = e + (lane>=16 ? 8 : 0), col = lane&15
  const int rb = (lane >> 4) * 8;
  const int cn = lane & 15;
#pragma unroll
  for (int i = 0; i < 4; ++i)
#pragma unroll
    for (int j = 0; j < 4; ++j) {
      const long mb = wm + i * 16 + rb;
      const long col = wn + j * 16 + cn;
#pragma unroll
      for (int e = 0; e < 8; ++e) {
        const long m = mb + e;
        const float v = acc[i][j][e];
        if (MODE == 0)      ((_Float16*)Cout)[m * N + col] = (_Float16)v;
        else if (MODE == 1) ((_Float16*)Cout)[col * M + m] = (_Float16)v;
        else                ((float*)Cout)[m * N + col] = v;
      }
    }
}

// ---------------------------------------------------------------------------
// RoPE (NeoX split style, matching the reference):
//   out[j]    = x[2j]*cos - x[2j+1]*sin
//   out[j+64] = x[2j]*sin + x[2j+1]*cos       j in [0,64)
// One block (64 threads) per (token, head). Separate out buffer (no aliasing).
// ---------------------------------------------------------------------------
__global__ void rope_kernel(const _Float16* __restrict__ in,
                            _Float16* __restrict__ out, int nheads) {
  const int j = threadIdx.x;                    // 0..63
  const long row = blockIdx.x;                  // token*nheads + h
  const long token = row / nheads;
  const long pos = token % SS;
  const float freq = powf(10000.0f, -(float)j / 64.0f);
  const float ang = (float)pos * freq;
  const float c = cosf(ang), s = sinf(ang);
  const _Float16* p = in  + row * HEAD_DIM;
  _Float16*       o = out + row * HEAD_DIM;
  const float x1 = (float)p[2 * j], x2 = (float)p[2 * j + 1];
  o[j]      = (_Float16)(x1 * c - x2 * s);
  o[j + 64] = (_Float16)(x1 * s + x2 * c);
}

// ---------------------------------------------------------------------------
// Causal flash attention, one wave per 16-query tile.
//   Q: [token, h*128+d] f16 (post-RoPE)   K: [token, kv*128+d] f16 (post-RoPE)
//   Vt: [kv*128+d, token] f16 (pre-transposed by gemm MODE 1)
//   O: [token, h*128+d] f16
// Key blocks of 32 -> P tile is one 16x32 A-fragment for the P@V WMMA.
// Row stats kept per lane for the 8 rows it owns in the C layout; row
// reductions are 16-lane shfl_xor butterflies (rows live in wave halves).
// P is re-strided C-layout -> A-fragment through a padded LDS tile.
// ---------------------------------------------------------------------------
__global__ __launch_bounds__(256) void attn_flash(
    const _Float16* __restrict__ Q, const _Float16* __restrict__ Kr,
    const _Float16* __restrict__ Vt, _Float16* __restrict__ O) {
  const int wave = blockIdx.x * 8 + (threadIdx.x >> 5);
  const int qtiles = SS / 16;                    // 128
  const int qt = wave % qtiles;
  const int bh = wave / qtiles;
  const int h = bh % N_HEADS;
  const int b = bh / N_HEADS;
  if (b >= BB) return;                           // wave-uniform
  const int kvh = h & (N_KV - 1);                // jnp.tile -> kv = h % 4
  const int lane = threadIdx.x & 31;
  const int w = (threadIdx.x >> 5) & 7;

  __shared__ _Float16 plds[8][16][40];           // padded: distinct banks
  const _Float16* Qp = Q  + (long)b * SS * QDIM  + (long)h   * HEAD_DIM;
  const _Float16* Kp = Kr + (long)b * SS * KVDIM + (long)kvh * HEAD_DIM;
  const _Float16* Vp = Vt + (long)kvh * HEAD_DIM * TOK + (long)b * SS;

  v16h qf[4];
#pragma unroll
  for (int kk = 0; kk < 4; ++kk)
    qf[kk] = frag_rowmajor(Qp, QDIM, (long)qt * 16, kk * 32);

  v8f oacc[8] = {};                              // 16q x 128d accumulator
  float mrow[8], lrow[8];
#pragma unroll
  for (int e = 0; e < 8; ++e) { mrow[e] = -1e30f; lrow[e] = 0.0f; }

  const int rb = (lane >> 4) * 8;
  const int cn = lane & 15;
  const float scale = 0.088388347648318447f;     // 1/sqrt(128)

  const int nkb = ((qt * 16 + 16) + 31) >> 5;    // causal: ceil((qmax+1)/32)
  for (int kb = 0; kb < nkb; ++kb) {
    const int k0 = kb * 32;

    // ---- scores: 16q x 32k = two C tiles, K dim 128 -> 8 WMMAs -------------
    v8f s0 = {}, s1 = {};
#pragma unroll
    for (int kk = 0; kk < 4; ++kk) {
      v16h kf0 = frag_rowmajor(Kp, KVDIM, (long)k0,      kk * 32);
      v16h kf1 = frag_rowmajor(Kp, KVDIM, (long)k0 + 16, kk * 32);
      s0 = wmma16(qf[kk], kf0, s0);
      s1 = wmma16(qf[kk], kf1, s1);
    }

    // ---- online softmax ----------------------------------------------------
#pragma unroll
    for (int e = 0; e < 8; ++e) {
      const int qpos = qt * 16 + rb + e;
      float v0 = s0[e] * scale;
      float v1 = s1[e] * scale;
      if (k0 + cn      > qpos) v0 = -1e30f;      // causal mask
      if (k0 + 16 + cn > qpos) v1 = -1e30f;
      float mx = fmaxf(v0, v1);
#pragma unroll
      for (int off = 1; off < 16; off <<= 1)     // row max across 16-lane half
        mx = fmaxf(mx, __shfl_xor(mx, off, 32));
      const float mnew = fmaxf(mrow[e], mx);
      const float p0 = __expf(v0 - mnew);
      const float p1 = __expf(v1 - mnew);
      float rs = p0 + p1;
#pragma unroll
      for (int off = 1; off < 16; off <<= 1)     // row sum across half
        rs += __shfl_xor(rs, off, 32);
      const float corr = __expf(mrow[e] - mnew);
      lrow[e] = lrow[e] * corr + rs;
      mrow[e] = mnew;
#pragma unroll
      for (int j = 0; j < 8; ++j) oacc[j][e] *= corr;
      plds[w][rb + e][cn]      = (_Float16)p0;   // stage P for A-fragment
      plds[w][rb + e][16 + cn] = (_Float16)p1;
    }
    // same-wave LDS RAW: wait on the CDNA5 split DS counter
    asm volatile("s_wait_dscnt 0x0" ::: "memory");

    // ---- P (16x32) @ V (32x128): 8 WMMAs -----------------------------------
    v16h pf = frag_rowmajor(&plds[w][0][0], 40, 0, 0);
#pragma unroll
    for (int j = 0; j < 8; ++j) {
      v16h vf = frag_rowmajor(Vp, TOK, (long)j * 16, (long)k0);
      oacc[j] = wmma16(pf, vf, oacc[j]);
    }
  }

  // ---- normalize + store ----------------------------------------------------
  _Float16* Op = O + ((long)b * SS + (long)qt * 16) * QDIM + (long)h * HEAD_DIM;
#pragma unroll
  for (int e = 0; e < 8; ++e) {
    const float inv = 1.0f / lrow[e];
    const long r = rb + e;
#pragma unroll
    for (int j = 0; j < 8; ++j)
      Op[r * QDIM + j * 16 + cn] = (_Float16)(oacc[j][e] * inv);
  }
}

// ---------------------------------------------------------------------------
// Host side
// ---------------------------------------------------------------------------
extern "C" void kernel_launch(void* const* d_in, const int* in_sizes, int n_in,
                              void* d_out, int out_size, void* d_ws, size_t ws_size,
                              hipStream_t stream) {
  const float* x  = (const float*)d_in[0];
  const float* wq = (const float*)d_in[1];
  const float* wk = (const float*)d_in[2];
  const float* wv = (const float*)d_in[3];
  const float* wo = (const float*)d_in[4];
  // d_in[5] = causal mask, applied analytically in attn_flash
  float* out = (float*)d_out;

  char* ws = (char*)d_ws;
  _Float16* xh  = (_Float16*)ws; ws += (long)TOK * EMB * 2;      // 16 MB
  _Float16* wqT = (_Float16*)ws; ws += (long)QDIM * EMB * 2;     //  8 MB  [N,K]
  _Float16* wkT = (_Float16*)ws; ws += (long)KVDIM * EMB * 2;    //  2 MB
  _Float16* wvT = (_Float16*)ws; ws += (long)KVDIM * EMB * 2;    //  2 MB
  _Float16* woT = (_Float16*)ws; ws += (long)EMB * QDIM * 2;     //  8 MB
  _Float16* Qh  = (_Float16*)ws; ws += (long)TOK * QDIM * 2;     // 16 MB
  _Float16* Kh  = (_Float16*)ws; ws += (long)TOK * KVDIM * 2;    //  4 MB
  _Float16* Vt  = (_Float16*)ws; ws += (long)KVDIM * TOK * 2;    //  4 MB  [N,M]
  _Float16* Qr  = (_Float16*)ws; ws += (long)TOK * QDIM * 2;     // 16 MB
  _Float16* Kr  = (_Float16*)ws; ws += (long)TOK * KVDIM * 2;    //  4 MB
  _Float16* Oh  = (_Float16*)ws; ws += (long)TOK * QDIM * 2;     // 16 MB

  // 1) precision conversion + weight transposition (B^T layout for WMMA)
  {
    long n = (long)TOK * EMB;
    cvt_f32_to_f16<<<(n + 255) / 256, 256, 0, stream>>>(x, xh, n);
    long nq = (long)QDIM * EMB;
    transpose_cvt<<<(nq + 255) / 256, 256, 0, stream>>>(wq, wqT, EMB, QDIM);
    long nk = (long)KVDIM * EMB;
    transpose_cvt<<<(nk + 255) / 256, 256, 0, stream>>>(wk, wkT, EMB, KVDIM);
    transpose_cvt<<<(nk + 255) / 256, 256, 0, stream>>>(wv, wvT, EMB, KVDIM);
    transpose_cvt<<<(nq + 255) / 256, 256, 0, stream>>>(wo, woT, QDIM, EMB);
  }

  // 2) QKV projections (WMMA). V stored transposed for the P@V B-operand.
  {
    int qb = ((TOK / 64) * (QDIM / 64)) / 8;            // 256 blocks
    gemm_wmma<0><<<qb, 256, 0, stream>>>(xh, wqT, Qh, TOK, QDIM, EMB);
    int kb = ((TOK / 64) * (KVDIM / 64)) / 8;           // 64 blocks
    gemm_wmma<0><<<kb, 256, 0, stream>>>(xh, wkT, Kh, TOK, KVDIM, EMB);
    gemm_wmma<1><<<kb, 256, 0, stream>>>(xh, wvT, Vt, TOK, KVDIM, EMB);
  }

  // 3) RoPE on Q and K
  rope_kernel<<<(long)TOK * N_HEADS, 64, 0, stream>>>(Qh, Qr, N_HEADS);
  rope_kernel<<<(long)TOK * N_KV,    64, 0, stream>>>(Kh, Kr, N_KV);

  // 4) causal flash attention: B*H*(S/16) = 4096 waves, 8 waves/block
  attn_flash<<<(BB * N_HEADS * (SS / 16)) / 8, 256, 0, stream>>>(Qr, Kr, Vt, Oh);

  // 5) output projection, f32 result straight into d_out
  {
    int ob = ((TOK / 64) * (EMB / 64)) / 8;             // 256 blocks
    gemm_wmma<2><<<ob, 256, 0, stream>>>(Oh, woT, out, TOK, EMB, QDIM);
  }
}
